// GenericEdgeAttrHeteroConv_52922587021516
// MI455X (gfx1250) — matrix operand order repaired and checked by reading, not compiled, not run on hardware
//
#include <hip/hip_runtime.h>
#include <math.h>

// ---------------------------------------------------------------------------
// MI455X / gfx1250 implementation of GenericEdgeAttrHeteroConv forward.
// wave32, WMMA f32<-f16 16x16x32 for dense math, async global->LDS staging
// for (pre-transposed, pre-converted) weight tiles.
// ---------------------------------------------------------------------------

typedef __attribute__((ext_vector_type(16))) _Float16 v16h;
typedef __attribute__((ext_vector_type(8)))  _Float16 v8h;
typedef __attribute__((ext_vector_type(4)))  _Float16 v4h;
typedef __attribute__((ext_vector_type(8)))  float    v8f;

#define N_DRUG 100000
#define N_DIS  50000
#define E1N    500000
#define E2N    250000

// monotone float <-> uint32 key for atomicMax-based segment max
__device__ __forceinline__ unsigned f2ord(float f) {
    unsigned b = __float_as_uint(f);
    return (b & 0x80000000u) ? ~b : (b | 0x80000000u);
}
__device__ __forceinline__ float ord2f(unsigned k) {
    unsigned b = (k & 0x80000000u) ? (k ^ 0x80000000u) : ~k;
    return __uint_as_float(b);
}

// Async copy of 16 bytes global -> LDS (ASYNCcnt-tracked, no VGPR transit).
// vdst operand = wave-relative LDS byte address (low 32 bits of generic ptr).
__device__ __forceinline__ void async_copy16(const void* gsrc, void* lds) {
    unsigned ldsOff = (unsigned)(size_t)lds;
    asm volatile("global_load_async_to_lds_b128 %0, %1, off"
                 :: "v"(ldsOff), "v"(gsrc) : "memory");
}
__device__ __forceinline__ void async_wait0() {
    asm volatile("s_wait_asynccnt 0x0" ::: "memory");
}

// ---------------------------------------------------------------------------
// One-time weight prep: transpose 128x128 f32 W (row-major [k][n]) into
// f16 W^T ([n][k]) so GEMM blocks can stage it with pure async byte copies.
// ---------------------------------------------------------------------------
__global__ __launch_bounds__(256) void prep_wt128(const float* __restrict__ W,
                                                  _Float16* __restrict__ Wt)
{
    int i = blockIdx.x * 256 + threadIdx.x;   // 16384
    int k = i >> 7, c = i & 127;
    Wt[c * 128 + k] = (_Float16)W[i];
}

// gW1 [36,128] -> f16 W1^T [128 n][64 k], K zero-padded 36 -> 64
__global__ __launch_bounds__(256) void prep_w1t(const float* __restrict__ gW1,
                                                _Float16* __restrict__ W1t)
{
    int i = blockIdx.x * 256 + threadIdx.x;   // 8192
    int c = i >> 6, k = i & 63;
    W1t[i] = (k < 36) ? (_Float16)gW1[k * 128 + c] : (_Float16)0.f;
}

// ---------------------------------------------------------------------------
// C[M,128] = f16(A[M,128]) @ W + bias, W given pre-transposed/converted f16.
// Block = 256 thr = 8 waves; block tile 128 rows; wave tile 16 rows x 128 cols.
// A/B fragments: contiguous k (16*half + s), not the ISA 7.12.2 table order; f32 C 16x16 per ISA.
// ---------------------------------------------------------------------------
__global__ __launch_bounds__(256) void gemm128(const float* __restrict__ A,
                                               const _Float16* __restrict__ Wt,
                                               const float* __restrict__ bias,
                                               float* __restrict__ C, int M)
{
    __shared__ __align__(16) _Float16 sA[128 * 128];   // 32 KB
    __shared__ __align__(16) _Float16 sWt[128 * 128];  // 32 KB  [n][k]
    const int tid = threadIdx.x;
    const int rowBase = blockIdx.x * 128;

    // W^T tile: pure async byte copy (2048 x 16B chunks)
    for (int c = tid; c < 2048; c += 256)
        async_copy16((const char*)Wt + (size_t)c * 16, (char*)sWt + c * 16);
    // A tile: convert f32 -> f16 (row guard zero-fills the tail block)
    for (int i = tid; i < 128 * 128; i += 256) {
        int r = i >> 7, c = i & 127;
        int gr = rowBase + r;
        sA[i] = (gr < M) ? (_Float16)A[(size_t)gr * 128 + c] : (_Float16)0.f;
    }
    async_wait0();
    __syncthreads();

    const int wave = tid >> 5, lane = tid & 31;
    const int lrow = lane & 15, half = lane >> 4;
    const int r = wave * 16 + lrow;

    v8f acc[8] = {};
#pragma unroll
    for (int k0 = 0; k0 < 128; k0 += 32) {
        v8h lo = *(const v8h*)&sA[r * 128 + k0 + half * 16];
        v8h hi = *(const v8h*)&sA[r * 128 + k0 + half * 16 + 8];
        v16h af = __builtin_shufflevector(lo, hi, 0, 1, 2, 3, 4, 5, 6, 7,
                                          8, 9, 10, 11, 12, 13, 14, 15);
        v16h bf[8];
#pragma unroll
        for (int n = 0; n < 8; ++n)      // preload: 16 back-to-back ds_load_b128
            bf[n] = *(const v16h*)&sWt[(n * 16 + lrow) * 128 + k0 + half * 16];
#pragma unroll
        for (int n = 0; n < 8; ++n)      // WMMA burst
            acc[n] = __builtin_amdgcn_wmma_f32_16x16x32_f16(
                false, af, false, bf[n], (short)0, acc[n], false, false);
    }
#pragma unroll
    for (int n = 0; n < 8; ++n) {
#pragma unroll
        for (int rr = 0; rr < 8; ++rr) {
            int grow = rowBase + wave * 16 + half * 8 + rr;
            int gcol = n * 16 + lrow;
            if (grow < M)
                C[(size_t)grow * 128 + gcol] = acc[n][rr] + bias[gcol];
        }
    }
}

// ---------------------------------------------------------------------------
// Fused per-edge gate MLP:
// gate = sigmoid(gelu(concat(emb1[cat], cont) @ gW1 + gb1) @ gW2 + gb2)
// 128 edges / block, K1 padded 36 -> 64, output f16.
// LDS: [featA 16KB | W1t 16KB | W2t 32KB]; featA+W1t reused as midA (32KB).
// ---------------------------------------------------------------------------
__global__ __launch_bounds__(256) void gate_mlp(
    const float* __restrict__ emb1, const float* __restrict__ cont,
    const int* __restrict__ cat,
    const _Float16* __restrict__ W1t_g,   // [128 n][64 k] f16, prepped
    const float* __restrict__ gb1,
    const _Float16* __restrict__ W2t_g,   // [128 n][128 k] f16, prepped
    const float* __restrict__ gb2,
    _Float16* __restrict__ gate, int E)
{
    __shared__ __align__(16) _Float16 smem[32768];   // 64 KB
    _Float16* featA = smem;                          // [128][64]
    _Float16* W1t   = smem + 8192;                   // [128 n][64 k]
    _Float16* midA  = smem;                          // [128][128] (after barrier)
    _Float16* W2t   = smem + 16384;                  // [128 n][128 k]

    const int tid = threadIdx.x;
    const int eBase = blockIdx.x * 128;

    // async-stage both weight tiles (1024 + 2048 x 16B chunks)
    for (int c = tid; c < 1024; c += 256)
        async_copy16((const char*)W1t_g + (size_t)c * 16, (char*)W1t + c * 16);
    for (int c = tid; c < 2048; c += 256)
        async_copy16((const char*)W2t_g + (size_t)c * 16, (char*)W2t + c * 16);
    // gather feature rows (categorical embedding + continuous, pad to 64)
    for (int i = tid; i < 8192; i += 256) {
        int rrow = i >> 6, col = i & 63;
        int e = eBase + rrow;
        float v = 0.f;
        if (e < E) {
            if (col < 32)      v = emb1[cat[e] * 32 + col];
            else if (col < 36) v = cont[(size_t)e * 4 + (col - 32)];
        }
        featA[i] = (_Float16)v;
    }
    async_wait0();
    __syncthreads();

    const int wave = tid >> 5, lane = tid & 31;
    const int lrow = lane & 15, half = lane >> 4;
    const int r = wave * 16 + lrow;

    v8f acc[8] = {};
#pragma unroll
    for (int k0 = 0; k0 < 64; k0 += 32) {            // layer 1, K = 64
        v8h lo = *(const v8h*)&featA[r * 64 + k0 + half * 16];
        v8h hi = *(const v8h*)&featA[r * 64 + k0 + half * 16 + 8];
        v16h af = __builtin_shufflevector(lo, hi, 0, 1, 2, 3, 4, 5, 6, 7,
                                          8, 9, 10, 11, 12, 13, 14, 15);
        v16h bf[8];
#pragma unroll
        for (int n = 0; n < 8; ++n)
            bf[n] = *(const v16h*)&W1t[(n * 16 + lrow) * 64 + k0 + half * 16];
#pragma unroll
        for (int n = 0; n < 8; ++n)
            acc[n] = __builtin_amdgcn_wmma_f32_16x16x32_f16(
                false, af, false, bf[n], (short)0, acc[n], false, false);
    }
    __syncthreads();                                 // done reading featA/W1t
#pragma unroll
    for (int n = 0; n < 8; ++n) {                    // exact GELU -> midA (f16)
#pragma unroll
        for (int rr = 0; rr < 8; ++rr) {
            int mrow = wave * 16 + half * 8 + rr;
            int mcol = n * 16 + lrow;
            float x = acc[n][rr] + gb1[mcol];
            float g = 0.5f * x * (1.0f + erff(x * 0.70710678118654752f));
            midA[mrow * 128 + mcol] = (_Float16)g;
        }
    }
    __syncthreads();

#pragma unroll
    for (int n = 0; n < 8; ++n) acc[n] = (v8f){};
#pragma unroll
    for (int k0 = 0; k0 < 128; k0 += 32) {           // layer 2, K = 128
        v8h lo = *(const v8h*)&midA[r * 128 + k0 + half * 16];
        v8h hi = *(const v8h*)&midA[r * 128 + k0 + half * 16 + 8];
        v16h af = __builtin_shufflevector(lo, hi, 0, 1, 2, 3, 4, 5, 6, 7,
                                          8, 9, 10, 11, 12, 13, 14, 15);
        v16h bf[8];
#pragma unroll
        for (int n = 0; n < 8; ++n)
            bf[n] = *(const v16h*)&W2t[(n * 16 + lrow) * 128 + k0 + half * 16];
#pragma unroll
        for (int n = 0; n < 8; ++n)
            acc[n] = __builtin_amdgcn_wmma_f32_16x16x32_f16(
                false, af, false, bf[n], (short)0, acc[n], false, false);
    }
#pragma unroll
    for (int n = 0; n < 8; ++n) {                    // sigmoid -> global f16
#pragma unroll
        for (int rr = 0; rr < 8; ++rr) {
            int e = eBase + wave * 16 + half * 8 + rr;
            int col = n * 16 + lrow;
            if (e < E) {
                float x = acc[n][rr] + gb2[col];
                gate[(size_t)e * 128 + col] = (_Float16)(1.f / (1.f + expf(-x)));
            }
        }
    }
}

// ---------------------------------------------------------------------------
// Pass A: wave per edge. logits[e,h] = (sh.dh)/sqrt(32) + (mh.attn[h]),
// atomicMax per (dst, head) via ordered-uint keys. Lane owns 4 cols (b128).
// ---------------------------------------------------------------------------
__global__ __launch_bounds__(256) void edge_logits(
    const float* __restrict__ Hs, const float* __restrict__ Hd,
    const _Float16* __restrict__ gate,
    const int* __restrict__ si, const int* __restrict__ di,
    const float* __restrict__ attn, float* __restrict__ logits,
    unsigned* __restrict__ maxkey, int E)
{
    int e = blockIdx.x * 8 + (threadIdx.x >> 5);
    if (e >= E) return;
    int lane = threadIdx.x & 31;
    int s = si[e], d = di[e];
    float4 hs = *(const float4*)&Hs[(size_t)s * 128 + lane * 4];
    float4 hd = *(const float4*)&Hd[(size_t)d * 128 + lane * 4];
    float mx = hs.x + hd.x, my = hs.y + hd.y, mz = hs.z + hd.z, mw = hs.w + hd.w;
    if (gate) {
        v4h g = *(const v4h*)&gate[(size_t)e * 128 + lane * 4];
        mx *= (float)g[0]; my *= (float)g[1]; mz *= (float)g[2]; mw *= (float)g[3];
    }
    float4 a = *(const float4*)&attn[lane * 4];     // attn flat [4][32] == col
    float part = (hs.x * hd.x + hs.y * hd.y + hs.z * hd.z + hs.w * hd.w)
                     * 0.17677669529663689f         // 1/sqrt(32)
               + (mx * a.x + my * a.y + mz * a.z + mw * a.w);
    part += __shfl_xor(part, 1);                    // reduce within 8-lane head
    part += __shfl_xor(part, 2);
    part += __shfl_xor(part, 4);
    if ((lane & 7) == 0) {
        int h = lane >> 3;
        logits[(size_t)e * 4 + h] = part;
        atomicMax(&maxkey[(size_t)d * 4 + h], f2ord(part));
    }
}

// ---------------------------------------------------------------------------
// Pass B: thread per (edge, head): ex = exp(l - max); sum via atomicAdd.
// ---------------------------------------------------------------------------
__global__ __launch_bounds__(256) void exp_sum(
    float* __restrict__ logits, const int* __restrict__ di,
    const unsigned* __restrict__ maxkey, float* __restrict__ sum, int E)
{
    int idx = blockIdx.x * 256 + threadIdx.x;
    if (idx >= E * 4) return;
    int e = idx >> 2, h = idx & 3;
    int d = di[e];
    float ex = expf(logits[idx] - ord2f(maxkey[(size_t)d * 4 + h]));
    logits[idx] = ex;
    atomicAdd(&sum[(size_t)d * 4 + h], ex);
}

// ---------------------------------------------------------------------------
// Pass C: wave per edge. Recompute msg (L2-resident gathers), scale by
// softmax weight, atomicAdd into aggr[dst].
// ---------------------------------------------------------------------------
__global__ __launch_bounds__(256) void scatter_msg(
    const float* __restrict__ Hs, const float* __restrict__ Hd,
    const _Float16* __restrict__ gate,
    const int* __restrict__ si, const int* __restrict__ di,
    const float* __restrict__ ex, const float* __restrict__ sum,
    float* __restrict__ aggr, int E)
{
    int e = blockIdx.x * 8 + (threadIdx.x >> 5);
    if (e >= E) return;
    int lane = threadIdx.x & 31;
    int s = si[e], d = di[e];
    int h = lane >> 3;
    float w = ex[(size_t)e * 4 + h] / (sum[(size_t)d * 4 + h] + 1e-16f);
    float4 hs = *(const float4*)&Hs[(size_t)s * 128 + lane * 4];
    float4 hd = *(const float4*)&Hd[(size_t)d * 128 + lane * 4];
    float mx = hs.x + hd.x, my = hs.y + hd.y, mz = hs.z + hd.z, mw = hs.w + hd.w;
    if (gate) {
        v4h g = *(const v4h*)&gate[(size_t)e * 128 + lane * 4];
        mx *= (float)g[0]; my *= (float)g[1]; mz *= (float)g[2]; mw *= (float)g[3];
    }
    float* dst = &aggr[(size_t)d * 128 + lane * 4];
    atomicAdd(dst + 0, mx * w);
    atomicAdd(dst + 1, my * w);
    atomicAdd(dst + 2, mz * w);
    atomicAdd(dst + 3, mw * w);
}

// ---------------------------------------------------------------------------
extern "C" void kernel_launch(void* const* d_in, const int* in_sizes, int n_in,
                              void* d_out, int out_size, void* d_ws, size_t ws_size,
                              hipStream_t stream)
{
    (void)in_sizes; (void)n_in; (void)out_size; (void)ws_size;
    const float* x_drug  = (const float*)d_in[0];
    const float* x_dis   = (const float*)d_in[1];
    const float* e1_cont = (const float*)d_in[2];
    const float* W_src1  = (const float*)d_in[3];
    const float* b_src1  = (const float*)d_in[4];
    const float* W_dst1  = (const float*)d_in[5];
    const float* b_dst1  = (const float*)d_in[6];
    const float* attn1   = (const float*)d_in[7];
    const float* emb1    = (const float*)d_in[8];
    const float* gW1     = (const float*)d_in[9];
    const float* gb1     = (const float*)d_in[10];
    const float* gW2     = (const float*)d_in[11];
    const float* gb2     = (const float*)d_in[12];
    const float* W_src2  = (const float*)d_in[13];
    const float* b_src2  = (const float*)d_in[14];
    const float* W_dst2  = (const float*)d_in[15];
    const float* b_dst2  = (const float*)d_in[16];
    const float* attn2   = (const float*)d_in[17];
    const float* Wo_drug = (const float*)d_in[18];
    const float* bo_drug = (const float*)d_in[19];
    const float* Wo_dis  = (const float*)d_in[20];
    const float* bo_dis  = (const float*)d_in[21];
    const int* e1_src = (const int*)d_in[22];
    const int* e1_dst = (const int*)d_in[23];
    const int* e2_src = (const int*)d_in[24];
    const int* e2_dst = (const int*)d_in[25];
    const int* e1_cat = (const int*)d_in[26];

    char* base = (char*)d_ws;
    size_t off = 0;
    auto take = [&](size_t bytes) -> char* {
        char* p = base + off;
        off = (off + bytes + 255) & ~(size_t)255;
        return p;
    };
    float*     Hsrc1 = (float*)take((size_t)N_DRUG * 128 * 4);
    float*     Hdst1 = (float*)take((size_t)N_DRUG * 128 * 4);
    float*     Hsrc2 = (float*)take((size_t)N_DRUG * 128 * 4);
    float*     Hdst2 = (float*)take((size_t)N_DIS  * 128 * 4);
    _Float16*  gateB = (_Float16*)take((size_t)E1N * 128 * 2);
    float*     lg1   = (float*)take((size_t)E1N * 4 * 4);
    float*     lg2   = (float*)take((size_t)E2N * 4 * 4);
    unsigned*  mk1   = (unsigned*)take((size_t)N_DRUG * 4 * 4);
    float*     sm1   = (float*)take((size_t)N_DRUG * 4 * 4);
    unsigned*  mk2   = (unsigned*)take((size_t)N_DIS * 4 * 4);
    float*     sm2   = (float*)take((size_t)N_DIS * 4 * 4);
    float*     aggD  = (float*)take((size_t)N_DRUG * 128 * 4);
    float*     aggS  = (float*)take((size_t)N_DIS  * 128 * 4);
    // pre-transposed / f16-converted weights (async-copyable tiles)
    _Float16*  WtS1  = (_Float16*)take(128 * 128 * 2);
    _Float16*  WtD1  = (_Float16*)take(128 * 128 * 2);
    _Float16*  WtS2  = (_Float16*)take(128 * 128 * 2);
    _Float16*  WtD2  = (_Float16*)take(128 * 128 * 2);
    _Float16*  WtOD  = (_Float16*)take(128 * 128 * 2);
    _Float16*  WtOS  = (_Float16*)take(128 * 128 * 2);
    _Float16*  Wt2g  = (_Float16*)take(128 * 128 * 2);
    _Float16*  Wt1g  = (_Float16*)take(128 * 64 * 2);

    hipMemsetAsync(mk1,  0, (size_t)N_DRUG * 4 * 4, stream);
    hipMemsetAsync(sm1,  0, (size_t)N_DRUG * 4 * 4, stream);
    hipMemsetAsync(mk2,  0, (size_t)N_DIS  * 4 * 4, stream);
    hipMemsetAsync(sm2,  0, (size_t)N_DIS  * 4 * 4, stream);
    hipMemsetAsync(aggD, 0, (size_t)N_DRUG * 128 * 4, stream);
    hipMemsetAsync(aggS, 0, (size_t)N_DIS  * 128 * 4, stream);

    const dim3 blk(256);
    const int gDrug = (N_DRUG + 127) / 128;
    const int gDis  = (N_DIS + 127) / 128;

    // weight prep (tiny)
    prep_wt128<<<64, blk, 0, stream>>>(W_src1,  WtS1);
    prep_wt128<<<64, blk, 0, stream>>>(W_dst1,  WtD1);
    prep_wt128<<<64, blk, 0, stream>>>(W_src2,  WtS2);
    prep_wt128<<<64, blk, 0, stream>>>(W_dst2,  WtD2);
    prep_wt128<<<64, blk, 0, stream>>>(Wo_drug, WtOD);
    prep_wt128<<<64, blk, 0, stream>>>(Wo_dis,  WtOS);
    prep_wt128<<<64, blk, 0, stream>>>(gW2,     Wt2g);
    prep_w1t <<<32, blk, 0, stream>>>(gW1,      Wt1g);

    // node transforms (WMMA, async W staging)
    gemm128<<<gDrug, blk, 0, stream>>>(x_drug, WtS1, b_src1, Hsrc1, N_DRUG);
    gemm128<<<gDrug, blk, 0, stream>>>(x_drug, WtD1, b_dst1, Hdst1, N_DRUG);
    gemm128<<<gDrug, blk, 0, stream>>>(x_drug, WtS2, b_src2, Hsrc2, N_DRUG);
    gemm128<<<gDis,  blk, 0, stream>>>(x_dis,  WtD2, b_dst2, Hdst2, N_DIS);

    // per-edge gate MLP (WMMA)
    gate_mlp<<<(E1N + 127) / 128, blk, 0, stream>>>(emb1, e1_cont, e1_cat,
                                                    Wt1g, gb1, Wt2g, gb2,
                                                    gateB, E1N);

    // segment softmax: logits+max, exp+sum, weighted scatter
    edge_logits<<<(E1N + 7) / 8, blk, 0, stream>>>(Hsrc1, Hdst1, gateB,
                                                   e1_src, e1_dst, attn1, lg1, mk1, E1N);
    edge_logits<<<(E2N + 7) / 8, blk, 0, stream>>>(Hsrc2, Hdst2, nullptr,
                                                   e2_src, e2_dst, attn2, lg2, mk2, E2N);
    exp_sum<<<(E1N * 4 + 255) / 256, blk, 0, stream>>>(lg1, e1_dst, mk1, sm1, E1N);
    exp_sum<<<(E2N * 4 + 255) / 256, blk, 0, stream>>>(lg2, e2_dst, mk2, sm2, E2N);
    scatter_msg<<<(E1N + 7) / 8, blk, 0, stream>>>(Hsrc1, Hdst1, gateB,
                                                   e1_src, e1_dst, lg1, sm1, aggD, E1N);
    scatter_msg<<<(E2N + 7) / 8, blk, 0, stream>>>(Hsrc2, Hdst2, nullptr,
                                                   e2_src, e2_dst, lg2, sm2, aggS, E2N);

    // output linears (WMMA) straight into d_out
    float* outD = (float*)d_out;
    float* outS = outD + (size_t)N_DRUG * 128;
    gemm128<<<gDrug, blk, 0, stream>>>(aggD, WtOD, bo_drug, outD, N_DRUG);
    gemm128<<<gDis,  blk, 0, stream>>>(aggS, WtOS, bo_dis,  outS, N_DIS);
}
